// GraphMLPMixer_30107720744961
// MI455X (gfx1250) — compile-verified, matrix operand order checked by reading, not run on hardware
//
#include <hip/hip_runtime.h>
#include <hip/hip_bf16.h>

// ---------------- problem constants (from reference) ----------------
#define D_    128
#define NOUT_ 10
#define P_    32
#define B_    64
#define TH_   64
#define CH_   512
#define N_    100000
#define M_    300000
#define E_    600000
#define E0_   400000
#define S_    2048
#define NFN_  64
#define NFE_  16
#define NLG_  4
#define NLM_  4

typedef __attribute__((ext_vector_type(16))) __bf16 v16bf;
typedef __attribute__((ext_vector_type(8)))  __bf16 v8bf;
typedef __attribute__((ext_vector_type(8)))  float  v8f;
typedef int i4v __attribute__((ext_vector_type(4)));

#if defined(__AMDGCN__) && __has_builtin(__builtin_amdgcn_global_load_async_to_lds_b128)
#define ASYNC_LDS 1
#else
#define ASYNC_LDS 0
#endif

__device__ __forceinline__ float gelu_f(float x) {
    float x3 = x * x * x;
    return 0.5f * x * (1.f + tanhf(0.7978845608028654f * (x + 0.044715f * x3)));
}

// =====================================================================
// WMMA GEMM:  OUT[Mrows,Ncols] = epi( (A (+Aadd)) @ Wt^T + bias )
// Wt pre-transposed bf16 [Ncols][Kpad]; bf16 in / f32 accumulate via
// v_wmma_f32_16x16x32_bf16.  Block 256 thr (8 waves), tile 128x128,
// BK=32, wave = 32x64 sub-tile (8 accumulators).
// B tiles staged with GLOBAL_LOAD_ASYNC_TO_LDS_B128 (ASYNCcnt);
// A tiles converted f32->bf16 (v_cvt_pk_bf16_f32) while staging.
// Requires: Kreal % 16 == 0, lda % 4 == 0 (true for all call sites).
// =====================================================================
#define BM 128
#define BN 128
#define BK 32
#define LDSTR 40   // 32 + 8 bf16 pad -> 80B row stride, bank-spread

template<bool HASADD, bool HASRES, int ACT>
__global__ __launch_bounds__(256)
void gemm_bf16(const float* __restrict__ A, const float* __restrict__ Aadd,
               const __bf16* __restrict__ Wt, const float* __restrict__ bias,
               const float* res, float* OUT,
               int Mrows, int Kpad, int Kreal, int lda, int Ncols)
{
    __shared__ __bf16 sA[BM * LDSTR];
    __shared__ __bf16 sB[BN * LDSTR];

    const int tid  = threadIdx.x;
    const int lane = tid & 31;
    const int wid  = tid >> 5;
    const int hl   = lane >> 4;
    const int r    = lane & 15;
    const int wm   = wid >> 1;      // 0..3 -> 32-row band
    const int wn   = wid & 1;       // 0..1 -> 64-col band

    const v8f zf = {0.f,0.f,0.f,0.f,0.f,0.f,0.f,0.f};
    v8f acc[2][4];
#pragma unroll
    for (int i = 0; i < 2; i++)
#pragma unroll
        for (int j = 0; j < 4; j++) acc[i][j] = zf;

    const int rowL  = tid >> 1;          // 0..127: staging row / col
    const int kOff  = (tid & 1) * 16;    // 0 or 16
    const int growA = blockIdx.x * BM + rowL;
    const int gcolB = blockIdx.y * BN + rowL;

    for (int kk = 0; kk < Kpad; kk += BK) {
        // ---- stage A tile: 4x float4 loads, mask-mult, 2x 16B LDS stores ----
        {
            const bool valid = (growA < Mrows) && (kk + kOff < Kreal);
            const size_t base = valid ? ((size_t)growA * (size_t)lda + kk + kOff) : 0;
            const float4* ap = (const float4*)(A + base);
            float4 f0 = ap[0], f1 = ap[1], f2 = ap[2], f3 = ap[3];
            if (HASADD) {
                const float4* cp = (const float4*)(Aadd + base);
                float4 c0 = cp[0], c1 = cp[1], c2 = cp[2], c3 = cp[3];
                f0.x += c0.x; f0.y += c0.y; f0.z += c0.z; f0.w += c0.w;
                f1.x += c1.x; f1.y += c1.y; f1.z += c1.z; f1.w += c1.w;
                f2.x += c2.x; f2.y += c2.y; f2.z += c2.z; f2.w += c2.w;
                f3.x += c3.x; f3.y += c3.y; f3.z += c3.z; f3.w += c3.w;
            }
            const float zm = valid ? 1.f : 0.f;
            v8bf lo, hi;
            lo[0] = (__bf16)(f0.x * zm); lo[1] = (__bf16)(f0.y * zm);
            lo[2] = (__bf16)(f0.z * zm); lo[3] = (__bf16)(f0.w * zm);
            lo[4] = (__bf16)(f1.x * zm); lo[5] = (__bf16)(f1.y * zm);
            lo[6] = (__bf16)(f1.z * zm); lo[7] = (__bf16)(f1.w * zm);
            hi[0] = (__bf16)(f2.x * zm); hi[1] = (__bf16)(f2.y * zm);
            hi[2] = (__bf16)(f2.z * zm); hi[3] = (__bf16)(f2.w * zm);
            hi[4] = (__bf16)(f3.x * zm); hi[5] = (__bf16)(f3.y * zm);
            hi[6] = (__bf16)(f3.z * zm); hi[7] = (__bf16)(f3.w * zm);
            *(v8bf*)&sA[rowL * LDSTR + kOff]     = lo;
            *(v8bf*)&sA[rowL * LDSTR + kOff + 8] = hi;
        }
        // ---- stage B tile: async global->LDS DMA (bf16 already) ----
        {
            const __bf16* wp = Wt + (size_t)gcolB * (size_t)Kpad + kk + kOff;
            __bf16* ldst = &sB[rowL * LDSTR + kOff];
#if ASYNC_LDS
            __builtin_amdgcn_global_load_async_to_lds_b128(
                (__attribute__((address_space(1))) i4v*)wp,
                (__attribute__((address_space(3))) i4v*)ldst, 0, 0);
            __builtin_amdgcn_global_load_async_to_lds_b128(
                (__attribute__((address_space(1))) i4v*)wp,
                (__attribute__((address_space(3))) i4v*)ldst, 16, 0);
#else
            *(v8bf*)ldst       = *(const v8bf*)wp;
            *(v8bf*)(ldst + 8) = *(const v8bf*)(wp + 8);
#endif
        }
#if ASYNC_LDS
#if __has_builtin(__builtin_amdgcn_s_wait_asynccnt)
        __builtin_amdgcn_s_wait_asynccnt(0);
#else
        asm volatile("s_wait_asynccnt 0" ::: "memory");
#endif
#endif
        __syncthreads();

        // ---- fragments (16-bit layout: lanes 0-15 K0-7/16-23, 16-31 K8-15/24-31) ----
        v16bf af[2], bfr[4];
#pragma unroll
        for (int tm = 0; tm < 2; tm++) {
            int row = wm * 32 + tm * 16 + r;
            v8bf lo = *(const v8bf*)&sA[row * LDSTR + hl * 8];
            v8bf hi = *(const v8bf*)&sA[row * LDSTR + 16 + hl * 8];
            af[tm] = __builtin_shufflevector(lo, hi, 0,1,2,3,4,5,6,7,8,9,10,11,12,13,14,15);
        }
#pragma unroll
        for (int tn = 0; tn < 4; tn++) {
            int col = wn * 64 + tn * 16 + r;
            v8bf lo = *(const v8bf*)&sB[col * LDSTR + hl * 8];
            v8bf hi = *(const v8bf*)&sB[col * LDSTR + 16 + hl * 8];
            bfr[tn] = __builtin_shufflevector(lo, hi, 0,1,2,3,4,5,6,7,8,9,10,11,12,13,14,15);
        }
#pragma unroll
        for (int tm = 0; tm < 2; tm++)
#pragma unroll
            for (int tn = 0; tn < 4; tn++)
                acc[tm][tn] = __builtin_amdgcn_wmma_f32_16x16x32_bf16(
                    false, af[tm], false, bfr[tn], (short)0, acc[tm][tn], false, false);
        __syncthreads();
    }

    // ---- epilogue: (vgpr rr, lane l): row = rr + 8*hl, col = l&15 ----
    // fullM: block-uniform branch -> straight-line stores for interior blocks;
    // output address strength-reduced (o += Ncols per row).
    const bool fullM = (blockIdx.x * BM + BM) <= (unsigned)Mrows;
#pragma unroll
    for (int tm = 0; tm < 2; tm++) {
        const int rowbase = blockIdx.x * BM + wm * 32 + tm * 16 + hl * 8;
#pragma unroll
        for (int tn = 0; tn < 4; tn++) {
            const int colg = blockIdx.y * BN + wn * 64 + tn * 16 + r;
            const float bv = bias[colg];
            size_t o = (size_t)rowbase * (size_t)Ncols + colg;
            if (fullM) {
#pragma unroll
                for (int rr = 0; rr < 8; rr++, o += Ncols) {
                    float v = acc[tm][tn][rr] + bv;
                    if (ACT == 1)      v = fmaxf(v, 0.f);
                    else if (ACT == 2) v = gelu_f(v);
                    if (HASRES) v += res[o];
                    OUT[o] = v;
                }
            } else {
#pragma unroll
                for (int rr = 0; rr < 8; rr++, o += Ncols) {
                    if (rowbase + rr < Mrows) {
                        float v = acc[tm][tn][rr] + bv;
                        if (ACT == 1)      v = fmaxf(v, 0.f);
                        else if (ACT == 2) v = gelu_f(v);
                        if (HASRES) v += res[o];
                        OUT[o] = v;
                    }
                }
            }
        }
    }
}

// ---------------- weight convert + transpose: Wt[n][k] = W[k][n] ----------------
__global__ void k_convwt(const float* __restrict__ w, __bf16* __restrict__ wt,
                         int K, int Nw, int Kpad)
{
    int n = blockIdx.x;
    for (int k = threadIdx.x; k < Kpad; k += blockDim.x)
        wt[(size_t)n * Kpad + k] = (k < K) ? (__bf16)w[(size_t)k * Nw + n] : (__bf16)0.f;
}

__global__ void k_zero(float* __restrict__ p, long long n)
{
    long long i  = (long long)blockIdx.x * blockDim.x + threadIdx.x;
    long long st = (long long)gridDim.x * blockDim.x;
    for (; i < n; i += st) p[i] = 0.f;
}

// ---------------- wave-per-row gather / scatter helpers (D=128) ----------------
__global__ void k_gather(float* __restrict__ out, const float* __restrict__ src,
                         const int* __restrict__ idx, int rows)
{
    int wv = (int)((blockIdx.x * (unsigned)blockDim.x + threadIdx.x) >> 5);
    int lane = threadIdx.x & 31;
    if (wv >= rows) return;
    const float4* s = (const float4*)(src + (size_t)idx[wv] * D_);
    float4* o = (float4*)(out + (size_t)wv * D_);
    o[lane] = s[lane];
}

__global__ void k_gather_add(float* __restrict__ out, const float* __restrict__ src,
                             const int* __restrict__ idx, int rows)
{
    int wv = (int)((blockIdx.x * (unsigned)blockDim.x + threadIdx.x) >> 5);
    int lane = threadIdx.x & 31;
    if (wv >= rows) return;
    const float* s = src + (size_t)idx[wv] * D_ + lane * 4;
    float* o = out + (size_t)wv * D_ + lane * 4;
#pragma unroll
    for (int j = 0; j < 4; j++) o[j] += s[j];
}

__global__ void k_gather_div(float* __restrict__ out, const float* __restrict__ acc,
                             const float* __restrict__ cnt, const int* __restrict__ idx,
                             int rows)
{
    int wv = (int)((blockIdx.x * (unsigned)blockDim.x + threadIdx.x) >> 5);
    int lane = threadIdx.x & 31;
    if (wv >= rows) return;
    int s = idx[wv];
    float inv = 1.f / fmaxf(cnt[s], 1.f);
    const float* a = acc + (size_t)s * D_ + lane * 4;
    float* o = out + (size_t)wv * D_ + lane * 4;
#pragma unroll
    for (int j = 0; j < 4; j++) o[j] = a[j] * inv;
}

__global__ void k_scatter_add(const float* __restrict__ x, const int* __restrict__ idx,
                              float* __restrict__ acc, float* __restrict__ cnt, int rows)
{
    int wv = (int)((blockIdx.x * (unsigned)blockDim.x + threadIdx.x) >> 5);
    int lane = threadIdx.x & 31;
    if (wv >= rows) return;
    int s = idx[wv];
    const float* xr = x + (size_t)wv * D_ + lane * 4;
    float* a = acc + (size_t)s * D_ + lane * 4;
#pragma unroll
    for (int j = 0; j < 4; j++) atomicAdd(&a[j], xr[j]);
    if (lane == 0) atomicAdd(&cnt[s], 1.f);
}

__global__ void k_div_rows(float* __restrict__ out, const float* __restrict__ acc,
                           const float* __restrict__ cnt, int rows)
{
    int wv = (int)((blockIdx.x * (unsigned)blockDim.x + threadIdx.x) >> 5);
    int lane = threadIdx.x & 31;
    if (wv >= rows) return;
    float inv = 1.f / fmaxf(cnt[wv], 1.f);
    const float* a = acc + (size_t)wv * D_ + lane * 4;
    float* o = out + (size_t)wv * D_ + lane * 4;
#pragma unroll
    for (int j = 0; j < 4; j++) o[j] = a[j] * inv;
}

// ---------------- GINE message + aggregate: agg[dst] += relu(x[src]+e) ----------------
__global__ void k_edge_msg(const float* __restrict__ x, const float* __restrict__ e0,
                           const int* __restrict__ src, const int* __restrict__ dst,
                           const int* __restrict__ emap, float* __restrict__ agg, int E)
{
    int e = (int)((blockIdx.x * (unsigned)blockDim.x + threadIdx.x) >> 5);
    int lane = threadIdx.x & 31;
    if (e >= E) return;
    int s = src[e], d = dst[e], ea = emap[e];
    const float* xs = x  + (size_t)s  * D_ + lane * 4;
    const float* es = e0 + (size_t)ea * D_ + lane * 4;
    float* ag = agg + (size_t)d * D_ + lane * 4;
#pragma unroll
    for (int j = 0; j < 4; j++) atomicAdd(&ag[j], fmaxf(xs[j] + es[j], 0.f));
}

// ---------------- LayerNorm over D=128 (wave per row) ----------------
__global__ void k_layernorm(const float* __restrict__ x, const float* __restrict__ g,
                            const float* __restrict__ b, float* __restrict__ y, int rows)
{
    int wv = (int)((blockIdx.x * (unsigned)blockDim.x + threadIdx.x) >> 5);
    int lane = threadIdx.x & 31;
    if (wv >= rows) return;
    const float* xr = x + (size_t)wv * D_ + lane * 4;
    float v0 = xr[0], v1 = xr[1], v2 = xr[2], v3 = xr[3];
    float s = v0 + v1 + v2 + v3;
    float q = v0*v0 + v1*v1 + v2*v2 + v3*v3;
#pragma unroll
    for (int o = 16; o > 0; o >>= 1) {
        s += __shfl_xor(s, o, 32);
        q += __shfl_xor(q, o, 32);
    }
    float mu = s * (1.f / D_);
    float var = q * (1.f / D_) - mu * mu;
    float rstd = rsqrtf(var + 1e-5f);
    float* yr = y + (size_t)wv * D_ + lane * 4;
    const float* gg = g + lane * 4;
    const float* bb = b + lane * 4;
    yr[0] = (v0 - mu) * rstd * gg[0] + bb[0];
    yr[1] = (v1 - mu) * rstd * gg[1] + bb[1];
    yr[2] = (v2 - mu) * rstd * gg[2] + bb[2];
    yr[3] = (v3 - mu) * rstd * gg[3] + bb[3];
}

// ---------------- token mixing: mx[b,p,d] += (gelu(yT@wt1+bt1)@wt2+bt2)[d,p] ----------------
__global__ __launch_bounds__(128)
void k_token_mix(const float* __restrict__ y, const float* __restrict__ wt1,
                 const float* __restrict__ bt1, const float* __restrict__ wt2,
                 const float* __restrict__ bt2, float* __restrict__ mx)
{
    __shared__ float ly[P_ * D_];
    int b = blockIdx.x, tid = threadIdx.x;
    for (int i = tid; i < P_ * D_; i += 128)
        ly[i] = y[(size_t)b * P_ * D_ + i];
    __syncthreads();
    int d = tid;
    float out[P_];
#pragma unroll
    for (int p = 0; p < P_; p++) out[p] = 0.f;
    for (int h = 0; h < TH_; h++) {
        float a = bt1[h];
#pragma unroll
        for (int p = 0; p < P_; p++) a += ly[p * D_ + d] * wt1[p * TH_ + h];
        a = gelu_f(a);
#pragma unroll
        for (int p = 0; p < P_; p++) out[p] += a * wt2[h * P_ + p];
    }
    float* m = mx + (size_t)b * P_ * D_;
#pragma unroll
    for (int p = 0; p < P_; p++) m[p * D_ + d] += out[p] + bt2[p];
}

// ---------------- readout head ----------------
__global__ __launch_bounds__(128)
void k_head(const float* __restrict__ mx, const unsigned char* __restrict__ mask,
            const float* __restrict__ w1, const float* __restrict__ b1,
            const float* __restrict__ w2, const float* __restrict__ b2,
            float* __restrict__ out)
{
    __shared__ float gx[D_];
    __shared__ float hh[D_];
    int b = blockIdx.x, d = threadIdx.x;
    float msum = 0.f;
#pragma unroll
    for (int p = 0; p < P_; p++) msum += mask[b * P_ + p] ? 1.f : 0.f;
    float acc = 0.f;
#pragma unroll
    for (int p = 0; p < P_; p++)
        if (mask[b * P_ + p]) acc += mx[((size_t)b * P_ + p) * D_ + d];
    gx[d] = acc / fmaxf(msum, 1.f);
    __syncthreads();
    float a = b1[d];
    for (int k = 0; k < D_; k++) a += gx[k] * w1[k * D_ + d];
    hh[d] = fmaxf(a, 0.f);
    __syncthreads();
    if (d < NOUT_) {
        float o = b2[d];
        for (int k = 0; k < D_; k++) o += hh[k] * w2[k * NOUT_ + d];
        out[b * NOUT_ + d] = o;
    }
}

// =====================================================================
extern "C" void kernel_launch(void* const* d_in, const int* in_sizes, int n_in,
                              void* d_out, int out_size, void* d_ws, size_t ws_size,
                              hipStream_t stream)
{
    (void)in_sizes; (void)n_in; (void)out_size; (void)ws_size;
    int p = 0;
    auto F = [&](int i) { return (const float*)d_in[i]; };
    auto I = [&](int i) { return (const int*)d_in[i]; };

    const float* x_feat       = F(p++);
    const float* edge_attr    = F(p++);
    const int*   nodes_mapper = I(p++);
    const int*   edges_mapper = I(p++);
    const int*   edge_index   = I(p++);
    const int*   batch_x      = I(p++);
    const unsigned char* mask = (const unsigned char*)d_in[p++];

    // params flattened in make_params insertion order
    const float* in_w = F(p++); const float* in_b = F(p++);
    const float* e_w  = F(p++); const float* e_b  = F(p++);
    const float *g_w1[NLG_], *g_b1[NLG_], *g_w2[NLG_], *g_b2[NLG_];
    for (int l = 0; l < NLG_; l++) { g_w1[l]=F(p++); g_b1[l]=F(p++); g_w2[l]=F(p++); g_b2[l]=F(p++); }
    const float *U_w[NLG_-1], *U_b[NLG_-1];
    for (int l = 0; l < NLG_-1; l++) { U_w[l]=F(p++); U_b[l]=F(p++); }
    const float *m_g1[NLM_], *m_bb1[NLM_], *m_wt1[NLM_], *m_bt1[NLM_], *m_wt2[NLM_], *m_bt2[NLM_],
                *m_g2[NLM_], *m_bb2[NLM_], *m_wc1[NLM_], *m_bc1[NLM_], *m_wc2[NLM_], *m_bc2[NLM_];
    for (int l = 0; l < NLM_; l++) {
        m_g1[l]=F(p++);  m_bb1[l]=F(p++); m_wt1[l]=F(p++); m_bt1[l]=F(p++);
        m_wt2[l]=F(p++); m_bt2[l]=F(p++); m_g2[l]=F(p++);  m_bb2[l]=F(p++);
        m_wc1[l]=F(p++); m_bc1[l]=F(p++); m_wc2[l]=F(p++); m_bc2[l]=F(p++);
    }
    const float* o_w1 = F(p++); const float* o_b1 = F(p++);
    const float* o_w2 = F(p++); const float* o_b2 = F(p++);

    const int* e_src = edge_index;        // edge_index[0]
    const int* e_dst = edge_index + E_;   // edge_index[1]

    // ---- carve workspace ----
    char* wsb = (char*)d_ws; size_t off = 0;
    auto alloc = [&](size_t bytes) { void* r = wsb + off; off += (bytes + 255) & ~(size_t)255; return r; };
    float* xN    = (float*)alloc((size_t)N_ * D_ * 4);   // encoder out, reused as node-sync acc
    float* cntN  = (float*)alloc((size_t)N_ * 4);
    float* e0    = (float*)alloc((size_t)E0_ * D_ * 4);
    float* xM    = (float*)alloc((size_t)M_ * D_ * 4);
    float* agg   = (float*)alloc((size_t)M_ * D_ * 4);
    float* hbuf  = (float*)alloc((size_t)M_ * D_ * 4);
    float* suba  = (float*)alloc((size_t)S_ * D_ * 4);
    float* cntS  = (float*)alloc((size_t)S_ * 4);
    float* sub   = (float*)alloc((size_t)S_ * D_ * 4);
    float* usub  = (float*)alloc((size_t)S_ * D_ * 4);
    float* mx    = (float*)alloc((size_t)S_ * D_ * 4);
    float* ybuf  = (float*)alloc((size_t)S_ * D_ * 4);
    float* h1    = (float*)alloc((size_t)S_ * CH_ * 4);
    __bf16* wt_in = (__bf16*)alloc(128 * 64 * 2);
    __bf16* wt_e  = (__bf16*)alloc(128 * 32 * 2);
    __bf16 *wt_g1[NLG_], *wt_g2[NLG_], *wt_U[NLG_-1], *wt_c1[NLM_], *wt_c2[NLM_];
    for (int l = 0; l < NLG_; l++)   { wt_g1[l] = (__bf16*)alloc(128*128*2); wt_g2[l] = (__bf16*)alloc(128*128*2); }
    for (int l = 0; l < NLG_-1; l++)   wt_U[l]  = (__bf16*)alloc(128*128*2);
    for (int l = 0; l < NLM_; l++)   { wt_c1[l] = (__bf16*)alloc((size_t)CH_*128*2); wt_c2[l] = (__bf16*)alloc((size_t)128*CH_*2); }

    dim3 blk(256);
    auto conv = [&](const float* w, __bf16* wt, int K, int Nw, int Kpad) {
        k_convwt<<<dim3((unsigned)Nw), dim3(128), 0, stream>>>(w, wt, K, Nw, Kpad);
    };
    auto zero = [&](float* ptr, long long n) {
        k_zero<<<dim3(2048), blk, 0, stream>>>(ptr, n);
    };
    auto wgrid = [](int rows) { return dim3((unsigned)((rows + 7) / 8)); }; // 8 waves/block
    auto ggrid = [](int Mr, int Nc) { return dim3((unsigned)((Mr + BM - 1) / BM), (unsigned)(Nc / BN)); };

    // ---- weight prep (tiny) ----
    conv(in_w, wt_in, NFN_, 128, 64);
    conv(e_w,  wt_e,  NFE_, 128, 32);
    for (int l = 0; l < NLG_; l++)  { conv(g_w1[l], wt_g1[l], 128, 128, 128); conv(g_w2[l], wt_g2[l], 128, 128, 128); }
    for (int l = 0; l < NLG_-1; l++)  conv(U_w[l], wt_U[l], 128, 128, 128);
    for (int l = 0; l < NLM_; l++)  { conv(m_wc1[l], wt_c1[l], 128, CH_, 128); conv(m_wc2[l], wt_c2[l], CH_, 128, CH_); }

    // ---- encoders ----
    gemm_bf16<false,false,0><<<ggrid(N_,128), blk, 0, stream>>>(
        x_feat, nullptr, wt_in, in_b, nullptr, xN, N_, 64, NFN_, NFN_, 128);
    gemm_bf16<false,false,0><<<ggrid(E0_,128), blk, 0, stream>>>(
        edge_attr, nullptr, wt_e, e_b, nullptr, e0, E0_, 32, NFE_, NFE_, 128);
    k_gather<<<wgrid(M_), blk, 0, stream>>>(xM, xN, nodes_mapper, M_);

    // ---- GNN layers ----
    for (int i = 0; i < NLG_; i++) {
        if (i > 0) {
            // patch mean -> U-MLP -> broadcast add
            zero(suba, (long long)S_ * D_); zero(cntS, S_);
            k_scatter_add<<<wgrid(M_), blk, 0, stream>>>(xM, batch_x, suba, cntS, M_);
            k_div_rows<<<wgrid(S_), blk, 0, stream>>>(sub, suba, cntS, S_);
            gemm_bf16<false,false,1><<<ggrid(S_,128), blk, 0, stream>>>(
                sub, nullptr, wt_U[i-1], U_b[i-1], nullptr, usub, S_, 128, 128, 128, 128);
            k_gather_add<<<wgrid(M_), blk, 0, stream>>>(xM, usub, batch_x, M_);
            // node-level mean sync
            zero(xN, (long long)N_ * D_); zero(cntN, N_);
            k_scatter_add<<<wgrid(M_), blk, 0, stream>>>(xM, nodes_mapper, xN, cntN, M_);
            k_gather_div<<<wgrid(M_), blk, 0, stream>>>(xM, xN, cntN, nodes_mapper, M_);
        }
        // GINE message + aggregate
        zero(agg, (long long)M_ * D_);
        k_edge_msg<<<wgrid(E_), blk, 0, stream>>>(xM, e0, e_src, e_dst, edges_mapper, agg, E_);
        // h = relu((x+agg)@w1+b1); x += relu(h@w2+b2)
        gemm_bf16<true,false,1><<<ggrid(M_,128), blk, 0, stream>>>(
            xM, agg, wt_g1[i], g_b1[i], nullptr, hbuf, M_, 128, 128, 128, 128);
        gemm_bf16<false,true,1><<<ggrid(M_,128), blk, 0, stream>>>(
            hbuf, nullptr, wt_g2[i], g_b2[i], xM, xM, M_, 128, 128, 128, 128);
    }

    // ---- patch pooling -> mixer tokens mx[S=B*P, D] ----
    zero(suba, (long long)S_ * D_); zero(cntS, S_);
    k_scatter_add<<<wgrid(M_), blk, 0, stream>>>(xM, batch_x, suba, cntS, M_);
    k_div_rows<<<wgrid(S_), blk, 0, stream>>>(mx, suba, cntS, S_);

    // ---- mixer layers ----
    for (int l = 0; l < NLM_; l++) {
        k_layernorm<<<wgrid(S_), blk, 0, stream>>>(mx, m_g1[l], m_bb1[l], ybuf, S_);
        k_token_mix<<<dim3(B_), dim3(128), 0, stream>>>(ybuf, m_wt1[l], m_bt1[l], m_wt2[l], m_bt2[l], mx);
        k_layernorm<<<wgrid(S_), blk, 0, stream>>>(mx, m_g2[l], m_bb2[l], ybuf, S_);
        gemm_bf16<false,false,2><<<ggrid(S_,CH_), blk, 0, stream>>>(
            ybuf, nullptr, wt_c1[l], m_bc1[l], nullptr, h1, S_, 128, 128, 128, CH_);
        gemm_bf16<false,true,0><<<ggrid(S_,128), blk, 0, stream>>>(
            h1, nullptr, wt_c2[l], m_bc2[l], mx, mx, S_, CH_, CH_, CH_, 128);
    }

    // ---- readout ----
    k_head<<<dim3(B_), dim3(128), 0, stream>>>(mx, mask, o_w1, o_b1, o_w2, o_b2, (float*)d_out);
}
